// LiteCrossFlowMSA_46849503265367
// MI455X (gfx1250) — matrix operand order repaired and checked by reading, not compiled, not run on hardware
//
#include <hip/hip_runtime.h>
#include <hip/hip_bf16.h>

// ---------------- problem constants ----------------
#define BQ    2
#define CINC  128
#define COUTC 128
#define HH    256
#define WWD   256
#define NN    (HH * WWD)      // 65536
#define GG    32              // attention groups (2*HEADS)
#define DD    8
#define EPSF  1e-6f

typedef __attribute__((ext_vector_type(16))) __bf16 v16bf;
typedef __attribute__((ext_vector_type(8)))  float  v8f;

union BfFrag {
    v16bf v;
    uint4 q[2];
    unsigned u[8];
};

__device__ __forceinline__ unsigned short f2bf(float f) {
    unsigned u = __float_as_uint(f);
    unsigned r = ((u >> 16) & 1u) + 0x7FFFu;   // round to nearest even
    return (unsigned short)((u + r) >> 16);
}
__device__ __forceinline__ float sigmoidf_(float x) { return 1.0f / (1.0f + __expf(-x)); }
__device__ __forceinline__ float wave_sum(float v) {
#pragma unroll
    for (int off = 16; off > 0; off >>= 1) v += __shfl_xor(v, off, 32);
    return v;
}
// Async copy one dword global -> LDS (gfx1250), tracked by ASYNCcnt.
__device__ __forceinline__ void async_g2l_b32(void* lds_ptr, const float* gptr) {
    unsigned ldsoff = (unsigned)(size_t)lds_ptr;
    asm volatile("global_load_async_to_lds_b32 %0, %1, off"
                 :: "v"(ldsoff), "v"(gptr) : "memory");
}

// ---------------- utility kernels ----------------
__global__ void zero_kernel(float* p, int count) {
    int i = blockIdx.x * 256 + threadIdx.x;
    if (i < count) p[i] = 0.f;
}

__global__ void convert_weights_kernel(const float* __restrict__ Wq,
                                       const float* __restrict__ Wkv,
                                       const float* __restrict__ Wp,
                                       unsigned short* __restrict__ wq,
                                       unsigned short* __restrict__ wkv,
                                       unsigned short* __restrict__ wp) {
    int i = blockIdx.x * 256 + threadIdx.x;
    if (i < 16384)      wq[i] = f2bf(Wq[i]);
    else if (i < 49152) wkv[i - 16384] = f2bf(Wkv[i - 16384]);
    else if (i < 81920) wp[i - 49152] = f2bf(Wp[i - 49152]);
}

// ---------------- 1x1 conv GEMM: out[b,o,n] = sum_i W[o,i]*x[b,i,n] ----------------
// 64-pixel tile per block; each wave: COUT/128 M-tiles x 4 N-subtiles, K=128 in 4 WMMA steps.
template <int COUT>
__global__ __launch_bounds__(256, 2) void gemm1x1_bf16(const float* __restrict__ x,
                                                       const unsigned short* __restrict__ Wbf,
                                                       float* __restrict__ out) {
    constexpr int MT = COUT / 128;            // M-tiles per wave (1 or 2)
    __shared__ unsigned short sIn[64][136];   // [pixel][K], padded vs bank conflicts
    const int b   = blockIdx.y;
    const int n0  = blockIdx.x * 64;
    const int tid = threadIdx.x;

    // stage + convert input tile (128 channels x 64 pixels) into LDS as bf16
    for (int i = tid; i < CINC * 64; i += 256) {
        int ci = i >> 6, n = i & 63;
        sIn[n][ci] = f2bf(x[((size_t)b * CINC + ci) * NN + n0 + n]);
    }
    __syncthreads();

    const int lane = tid & 31;
    const int wave = tid >> 5;
    const int m    = lane & 15;
    const bool hi  = lane >= 16;

    v8f c[MT][4];
#pragma unroll
    for (int mtl = 0; mtl < MT; ++mtl)
#pragma unroll
        for (int nt = 0; nt < 4; ++nt)
            c[mtl][nt] = (v8f){0.f, 0.f, 0.f, 0.f, 0.f, 0.f, 0.f, 0.f};

    const unsigned short* wrow[MT];
#pragma unroll
    for (int mtl = 0; mtl < MT; ++mtl)
        wrow[mtl] = Wbf + (size_t)((wave + 8 * mtl) * 16 + m) * CINC;

#pragma unroll
    for (int ks = 0; ks < CINC / 32; ++ks) {
        BfFrag a[MT];
        const int kbA = ks * 32 + (hi ? 8 : 0);
#pragma unroll
        for (int mtl = 0; mtl < MT; ++mtl) {
            a[mtl].q[0] = *reinterpret_cast<const uint4*>(wrow[mtl] + kbA);
            a[mtl].q[1] = *reinterpret_cast<const uint4*>(wrow[mtl] + kbA + 16);
        }
        const int kbB = ks * 32 + (hi ? 8 : 0);
#pragma unroll
        for (int nt = 0; nt < 4; ++nt) {
            BfFrag bb;
            bb.q[0] = *reinterpret_cast<const uint4*>(&sIn[nt * 16 + m][kbB]);
            bb.q[1] = *reinterpret_cast<const uint4*>(&sIn[nt * 16 + m][kbB + 16]);
#pragma unroll
            for (int mtl = 0; mtl < MT; ++mtl)
                c[mtl][nt] = __builtin_amdgcn_wmma_f32_16x16x32_bf16(
                    false, a[mtl].v, false, bb.v, (short)0, c[mtl][nt], false, false);
        }
    }
#pragma unroll
    for (int mtl = 0; mtl < MT; ++mtl) {
        const int mbase = (wave + 8 * mtl) * 16 + (hi ? 8 : 0);
#pragma unroll
        for (int nt = 0; nt < 4; ++nt)
#pragma unroll
            for (int r = 0; r < 8; ++r)
                out[((size_t)b * COUT + mbase + r) * NN + n0 + nt * 16 + m] = c[mtl][nt][r];
    }
}

// ---------------- q side: sigmoid(q0) and sigmoid(gconv(dwconv(q0))) ----------------
__global__ __launch_bounds__(256, 2) void qside_kernel(const float* __restrict__ q0,
                                                       const float* __restrict__ dwq,
                                                       const float* __restrict__ pwq,
                                                       float* __restrict__ qbuf,
                                                       float* __restrict__ qsum) {
    __shared__ float sT[8][20][21];
    __shared__ float sSum[16];
    const int b   = blockIdx.z;
    const int h   = blockIdx.y;
    const int ty0 = (blockIdx.x >> 4) * 16;
    const int tx0 = (blockIdx.x & 15) * 16;
    const int tid = threadIdx.x;
    if (tid < 16) sSum[tid] = 0.f;

    // zero tile (halo zero-padding), then async global->LDS copy of in-range region
    float* sTflat = &sT[0][0][0];
    for (int i = tid; i < 8 * 20 * 21; i += 256) sTflat[i] = 0.f;
    __syncthreads();
    for (int i = tid; i < 8 * 400; i += 256) {
        int cch = i / 400;
        int rem = i - cch * 400;
        int r = rem / 20, cc = rem - r * 20;
        int gy = ty0 - 2 + r, gx = tx0 - 2 + cc;
        if (gy >= 0 && gy < HH && gx >= 0 && gx < WWD)
            async_g2l_b32(&sT[cch][r][cc],
                          &q0[((size_t)b * 128 + h * 8 + cch) * NN + gy * WWD + gx]);
    }
    __builtin_amdgcn_s_wait_asynccnt(0);
    __syncthreads();

    const int py = tid >> 4, px = tid & 15;
    const int n  = (ty0 + py) * WWD + (tx0 + px);

    float qa[8], t[8], s1[8];
#pragma unroll
    for (int d = 0; d < 8; ++d) {
        qa[d] = sigmoidf_(sT[d][py + 2][px + 2]);
        float acc = 0.f;
#pragma unroll
        for (int i = 0; i < 5; ++i)
#pragma unroll
            for (int j = 0; j < 5; ++j)
                acc += dwq[(h * 8 + d) * 25 + i * 5 + j] * sT[d][py + i][px + j];
        t[d] = acc;
    }
    {
        float* dst = qbuf + ((size_t)(b * GG + h) * NN + n) * 8;
#pragma unroll
        for (int d = 0; d < 8; ++d) dst[d] = qa[d];
    }
#pragma unroll
    for (int e = 0; e < 8; ++e) {
        float acc = 0.f;
#pragma unroll
        for (int d = 0; d < 8; ++d) acc += pwq[(h * 8 + e) * 8 + d] * t[d];
        s1[e] = sigmoidf_(acc);
    }
    {
        float* dst = qbuf + ((size_t)(b * GG + 16 + h) * NN + n) * 8;
#pragma unroll
        for (int d = 0; d < 8; ++d) dst[d] = s1[d];
    }
#pragma unroll
    for (int d = 0; d < 8; ++d) {
        float v1 = wave_sum(qa[d]);
        float v2 = wave_sum(s1[d]);
        if ((tid & 31) == 0) { atomicAdd(&sSum[d], v1); atomicAdd(&sSum[8 + d], v2); }
    }
    __syncthreads();
    if (tid < 8)       atomicAdd(&qsum[(b * GG + h) * 8 + tid], sSum[tid]);
    else if (tid < 16) atomicAdd(&qsum[(b * GG + 16 + h) * 8 + (tid - 8)], sSum[tid]);
}

// ---------------- kv side ----------------
__global__ __launch_bounds__(256, 2) void kvside_kernel(const float* __restrict__ kv0,
                                                        const float* __restrict__ dwkv,
                                                        const float* __restrict__ pwkv,
                                                        float* __restrict__ kbuf,
                                                        float* __restrict__ vbuf,
                                                        float* __restrict__ ksum) {
    __shared__ float sT[16][20][21];
    __shared__ float sSum[8];
    const int b   = blockIdx.z;
    const int g   = blockIdx.y;
    const int ty0 = (blockIdx.x >> 4) * 16;
    const int tx0 = (blockIdx.x & 15) * 16;
    const int tid = threadIdx.x;
    if (tid < 8) sSum[tid] = 0.f;

    const int py = tid >> 4, px = tid & 15;
    const int n  = (ty0 + py) * WWD + (tx0 + px);
    float kk[8], vv[8];

    if (g < 16) {
        __syncthreads();   // sSum init visible
        const int cb = 16 * g;
#pragma unroll
        for (int d = 0; d < 8; ++d) {
            kk[d] = sigmoidf_(kv0[((size_t)b * 256 + cb + d) * NN + n]);
            vv[d] = kv0[((size_t)b * 256 + cb + 8 + d) * NN + n];
        }
    } else {
        const int mgrp = g - 16;
        const int cb   = 16 * mgrp;
        float* sTflat = &sT[0][0][0];
        for (int i = tid; i < 16 * 20 * 21; i += 256) sTflat[i] = 0.f;
        __syncthreads();
        for (int i = tid; i < 16 * 400; i += 256) {
            int cch = i / 400;
            int rem = i - cch * 400;
            int r = rem / 20, cc = rem - r * 20;
            int gy = ty0 - 2 + r, gx = tx0 - 2 + cc;
            if (gy >= 0 && gy < HH && gx >= 0 && gx < WWD)
                async_g2l_b32(&sT[cch][r][cc],
                              &kv0[((size_t)b * 256 + cb + cch) * NN + gy * WWD + gx]);
        }
        __builtin_amdgcn_s_wait_asynccnt(0);
        __syncthreads();

        float t[16];
#pragma unroll
        for (int c = 0; c < 16; ++c) {
            float acc = 0.f;
#pragma unroll
            for (int i = 0; i < 5; ++i)
#pragma unroll
                for (int j = 0; j < 5; ++j)
                    acc += dwkv[(cb + c) * 25 + i * 5 + j] * sT[c][py + i][px + j];
            t[c] = acc;
        }
#pragma unroll
        for (int e = 0; e < 8; ++e) {
            float ak = 0.f, av = 0.f;
#pragma unroll
            for (int d = 0; d < 8; ++d) {
                ak += pwkv[(2 * mgrp) * 64 + e * 8 + d] * t[d];
                av += pwkv[(2 * mgrp + 1) * 64 + e * 8 + d] * t[8 + d];
            }
            kk[e] = sigmoidf_(ak);
            vv[e] = av;
        }
    }

    {
        float* dk = kbuf + ((size_t)(b * GG + g) * NN + n) * 8;
        float* dv = vbuf + ((size_t)(b * GG + g) * NN + n) * 8;
#pragma unroll
        for (int d = 0; d < 8; ++d) { dk[d] = kk[d]; dv[d] = vv[d]; }
    }
#pragma unroll
    for (int d = 0; d < 8; ++d) {
        float v1 = wave_sum(kk[d]);
        if ((tid & 31) == 0) atomicAdd(&sSum[d], v1);
    }
    __syncthreads();
    if (tid < 8) atomicAdd(&ksum[(b * GG + g) * 8 + tid], sSum[tid]);
}

// ---------------- pass B: S_q = sum q*si, S_k = sum k*so ----------------
__global__ __launch_bounds__(256, 2) void passB_kernel(const float* __restrict__ qbuf,
                                                       const float* __restrict__ kbuf,
                                                       const float* __restrict__ qsum,
                                                       const float* __restrict__ ksum,
                                                       float* __restrict__ Sq,
                                                       float* __restrict__ Sk) {
    __shared__ float sA[16];
    const int b = blockIdx.z, g = blockIdx.y;
    const int tid = threadIdx.x;
    const int n = blockIdx.x * 256 + tid;
    if (tid < 16) sA[tid] = 0.f;
    __syncthreads();

    const float* qp = qbuf + ((size_t)(b * GG + g) * NN + n) * 8;
    const float* kp = kbuf + ((size_t)(b * GG + g) * NN + n) * 8;
    float q[8], k[8];
    float dsi = EPSF, dso = EPSF;
#pragma unroll
    for (int d = 0; d < 8; ++d) {
        q[d] = qp[d]; k[d] = kp[d];
        dsi += (q[d] + EPSF) * (ksum[(b * GG + g) * 8 + d] + EPSF);
        dso += (k[d] + EPSF) * (qsum[(b * GG + g) * 8 + d] + EPSF);
    }
    const float si = 1.f / dsi, so = 1.f / dso;
#pragma unroll
    for (int d = 0; d < 8; ++d) {
        float v1 = wave_sum(q[d] * si);
        float v2 = wave_sum(k[d] * so);
        if ((tid & 31) == 0) { atomicAdd(&sA[d], v1); atomicAdd(&sA[8 + d], v2); }
    }
    __syncthreads();
    if (tid < 8)       atomicAdd(&Sq[(b * GG + g) * 8 + tid], sA[tid]);
    else if (tid < 16) atomicAdd(&Sk[(b * GG + g) * 8 + tid - 8], sA[tid]);
}

// ---------------- pass C: softmax partition Z and unnormalized kvm matrix M ----------
__global__ __launch_bounds__(256, 2) void passC_kernel(const float* __restrict__ kbuf,
                                                       const float* __restrict__ vbuf,
                                                       const float* __restrict__ Sq,
                                                       float* __restrict__ Zbuf,
                                                       float* __restrict__ Mbuf) {
    __shared__ float sM[65];
    const int b = blockIdx.z, g = blockIdx.y;
    const int tid = threadIdx.x;
    const int n = blockIdx.x * 256 + tid;
    for (int i = tid; i < 65; i += 256) sM[i] = 0.f;
    __syncthreads();

    const float* kp = kbuf + ((size_t)(b * GG + g) * NN + n) * 8;
    const float* vp = vbuf + ((size_t)(b * GG + g) * NN + n) * 8;
    float k[8], v[8];
    float cs = EPSF;
#pragma unroll
    for (int d = 0; d < 8; ++d) {
        k[d] = kp[d]; v[d] = vp[d];
        cs += (k[d] + EPSF) * (Sq[(b * GG + g) * 8 + d] + EPSF);
    }
    cs = fminf(1.f, fmaxf(-1.f, cs));   // clip -> softmax needs no max pass
    const float wgt = __expf(cs);

    float wz = wave_sum(wgt);
    if ((tid & 31) == 0) atomicAdd(&sM[64], wz);
#pragma unroll
    for (int d = 0; d < 8; ++d)
#pragma unroll
        for (int e = 0; e < 8; ++e) {
            float p = wave_sum(wgt * k[d] * v[e]);
            if ((tid & 31) == 0) atomicAdd(&sM[d * 8 + e], p);
        }
    __syncthreads();
    if (tid < 64)       atomicAdd(&Mbuf[(b * GG + g) * 64 + tid], sM[tid]);
    else if (tid == 64) atomicAdd(&Zbuf[b * GG + g], sM[64]);
}

// ---------------- pass D: flow output + LN + Wp (WMMA) + BN -> d_out ----------------
// 32-pixel tile; wave = one 16-row M-tile over 2 N-subtiles, K=256 in 8 WMMA steps each.
__global__ __launch_bounds__(256, 2) void passD_kernel(const float* __restrict__ qbuf,
                                                       const unsigned short* __restrict__ wp_bf,
                                                       const float* __restrict__ ksum,
                                                       const float* __restrict__ Sk,
                                                       const float* __restrict__ Mbuf,
                                                       const float* __restrict__ Zbuf,
                                                       const float* __restrict__ ln_g,
                                                       const float* __restrict__ ln_b,
                                                       const float* __restrict__ bn_g,
                                                       const float* __restrict__ bn_b,
                                                       const float* __restrict__ bn_m,
                                                       const float* __restrict__ bn_v,
                                                       float* __restrict__ out) {
    __shared__ unsigned short sY[32][264];   // [pixel][K=256] bf16, padded
    __shared__ float sKVM[GG][64];
    const int b   = blockIdx.y;
    const int n0  = blockIdx.x * 32;
    const int tid = threadIdx.x;

    for (int i = tid; i < GG * 64; i += 256) {
        int g = i >> 6, idx = i & 63;
        sKVM[g][idx] = Mbuf[(b * GG + g) * 64 + idx] * ((float)NN / Zbuf[b * GG + g]);
    }
    __syncthreads();

    const int px = tid & 31;
#pragma unroll
    for (int gi = 0; gi < 4; ++gi) {
        const int g = (tid >> 5) * 4 + gi;
        const int n = n0 + px;
        const float* qp = qbuf + ((size_t)(b * GG + g) * NN + n) * 8;
        float q[8];
        float dsi = EPSF, csk = EPSF;
#pragma unroll
        for (int d = 0; d < 8; ++d) {
            q[d] = qp[d];
            dsi += (q[d] + EPSF) * (ksum[(b * GG + g) * 8 + d] + EPSF);
            csk += (q[d] + EPSF) * (Sk[(b * GG + g) * 8 + d] + EPSF);
        }
        const float scale = (1.f / dsi) * sigmoidf_(csk);   // ratio q.N/k.N == 1
        float o[8], mu = 0.f;
#pragma unroll
        for (int e = 0; e < 8; ++e) {
            float acc = 0.f;
#pragma unroll
            for (int d = 0; d < 8; ++d) acc += q[d] * sKVM[g][d * 8 + e];
            acc *= scale;
            o[e] = acc;
            mu += acc;
        }
        mu *= 0.125f;
        float var = 0.f;
#pragma unroll
        for (int e = 0; e < 8; ++e) { float dd = o[e] - mu; var += dd * dd; }
        var *= 0.125f;
        const float rs = rsqrtf(var + 1e-5f);
#pragma unroll
        for (int e = 0; e < 8; ++e) {
            float yn = (o[e] - mu) * rs * ln_g[e] + ln_b[e];
            sY[px][g * 8 + e] = f2bf(yn);
        }
    }
    __syncthreads();

    // WMMA: 256 -> 128 channels for 32 pixels
    const int lane = tid & 31;
    const int wave = tid >> 5;
    const int m    = lane & 15;
    const bool hi  = lane >= 16;
    v8f c[2];
    c[0] = (v8f){0.f, 0.f, 0.f, 0.f, 0.f, 0.f, 0.f, 0.f};
    c[1] = (v8f){0.f, 0.f, 0.f, 0.f, 0.f, 0.f, 0.f, 0.f};
    const unsigned short* wrow = wp_bf + (size_t)(wave * 16 + m) * 256;
#pragma unroll
    for (int ks = 0; ks < 8; ++ks) {
        BfFrag a;
        const int kbA = ks * 32 + (hi ? 8 : 0);
        a.q[0] = *reinterpret_cast<const uint4*>(wrow + kbA);
        a.q[1] = *reinterpret_cast<const uint4*>(wrow + kbA + 16);
        const int kbB = ks * 32 + (hi ? 8 : 0);
#pragma unroll
        for (int nt = 0; nt < 2; ++nt) {
            BfFrag bb;
            bb.q[0] = *reinterpret_cast<const uint4*>(&sY[nt * 16 + m][kbB]);
            bb.q[1] = *reinterpret_cast<const uint4*>(&sY[nt * 16 + m][kbB + 16]);
            c[nt] = __builtin_amdgcn_wmma_f32_16x16x32_bf16(false, a.v, false, bb.v,
                                                            (short)0, c[nt], false, false);
        }
    }
    const int mbase = wave * 16 + (hi ? 8 : 0);
#pragma unroll
    for (int nt = 0; nt < 2; ++nt)
#pragma unroll
        for (int r = 0; r < 8; ++r) {
            int ch = mbase + r;
            float val = (c[nt][r] - bn_m[ch]) * rsqrtf(bn_v[ch] + 1e-5f) * bn_g[ch] + bn_b[ch];
            out[((size_t)b * COUTC + ch) * NN + n0 + nt * 16 + m] = val;
        }
}

// ---------------- host side ----------------
extern "C" void kernel_launch(void* const* d_in, const int* in_sizes, int n_in,
                              void* d_out, int out_size, void* d_ws, size_t ws_size,
                              hipStream_t stream) {
    const float* x    = (const float*)d_in[0];
    const float* edge = (const float*)d_in[1];
    const float* Wq   = (const float*)d_in[2];
    const float* Wkv  = (const float*)d_in[3];
    const float* dwq  = (const float*)d_in[4];
    const float* pwq  = (const float*)d_in[5];
    const float* dwkv = (const float*)d_in[6];
    const float* pwkv = (const float*)d_in[7];
    const float* ln_g = (const float*)d_in[8];
    const float* ln_b = (const float*)d_in[9];
    const float* Wp   = (const float*)d_in[10];
    const float* bn_g = (const float*)d_in[11];
    const float* bn_b = (const float*)d_in[12];
    const float* bn_m = (const float*)d_in[13];
    const float* bn_v = (const float*)d_in[14];
    float* out = (float*)d_out;

    char* ws = (char*)d_ws;
    size_t off = 0;
    auto carve = [&](size_t bytes) -> char* {
        char* p = ws + off;
        off = (off + bytes + 255) & ~(size_t)255;
        return p;
    };
    unsigned short* wq_bf  = (unsigned short*)carve(128 * 128 * 2);
    unsigned short* wkv_bf = (unsigned short*)carve(256 * 128 * 2);
    unsigned short* wp_bf  = (unsigned short*)carve(128 * 256 * 2);
    float* stats = (float*)carve(6208 * 4);
    float* qsum = stats;           // 512
    float* ksum = stats + 512;     // 512
    float* Sq   = stats + 1024;    // 512
    float* Sk   = stats + 1536;    // 512
    float* Zbuf = stats + 2048;    // 64
    float* Mbuf = stats + 2112;    // 4096
    float* q0buf  = (float*)carve((size_t)BQ * 128 * NN * 4);
    float* kv0buf = (float*)carve((size_t)BQ * 256 * NN * 4);
    float* qbuf   = (float*)carve((size_t)BQ * GG * NN * 8 * 4);
    float* kbuf   = (float*)carve((size_t)BQ * GG * NN * 8 * 4);
    float* vbuf   = q0buf;   // alias: q0 is dead after qside_kernel runs

    zero_kernel<<<25, 256, 0, stream>>>(stats, 6208);
    convert_weights_kernel<<<320, 256, 0, stream>>>(Wq, Wkv, Wp, wq_bf, wkv_bf, wp_bf);

    gemm1x1_bf16<128><<<dim3(NN / 64, BQ), 256, 0, stream>>>(edge, wq_bf, q0buf);
    gemm1x1_bf16<256><<<dim3(NN / 64, BQ), 256, 0, stream>>>(x, wkv_bf, kv0buf);

    qside_kernel<<<dim3(256, 16, BQ), 256, 0, stream>>>(q0buf, dwq, pwq, qbuf, qsum);
    kvside_kernel<<<dim3(256, GG, BQ), 256, 0, stream>>>(kv0buf, dwkv, pwkv, kbuf, vbuf, ksum);

    passB_kernel<<<dim3(NN / 256, GG, BQ), 256, 0, stream>>>(qbuf, kbuf, qsum, ksum, Sq, Sk);
    passC_kernel<<<dim3(NN / 256, GG, BQ), 256, 0, stream>>>(kbuf, vbuf, Sq, Zbuf, Mbuf);

    passD_kernel<<<dim3(NN / 32, BQ), 256, 0, stream>>>(qbuf, wp_bf, ksum, Sk, Mbuf, Zbuf,
                                                        ln_g, ln_b, bn_g, bn_b, bn_m, bn_v, out);
}
